// LSTMtry_86887188398820
// MI455X (gfx1250) — compile-verified, matrix operand order checked by reading, not run on hardware
//
#include <hip/hip_runtime.h>
#include <hip/hip_bf16.h>
#include <math.h>

// Problem constants (match reference)
#define BB 64
#define TT 512
#define II 512
#define HH 1024
#define CC 1000
#define G4H (4 * HH)

typedef __attribute__((ext_vector_type(16))) __bf16 v16bf;
typedef __attribute__((ext_vector_type(8)))  float  v8f;
typedef __attribute__((ext_vector_type(4)))  int    i4v;

#define AS1 __attribute__((address_space(1)))
#define AS3 __attribute__((address_space(3)))

#if defined(__has_builtin)
#  if __has_builtin(__builtin_amdgcn_global_load_async_to_lds_b128)
#    define HAVE_ASYNC_LDS 1
#  endif
#endif
#ifndef HAVE_ASYNC_LDS
#  define HAVE_ASYNC_LDS 0
#endif

// ---------------------------------------------------------------------------
// WMMA helpers
// ---------------------------------------------------------------------------

// Load a 16x32 bf16 A/B fragment for v_wmma_f32_16x16x32_bf16.
// ISA layout (CDNA5 ISA): lane l holds row/col (l&15);
// vector element e maps to k = 16*(e>>3) + 8*(l>>4) + (e&7)
// => two contiguous 16-byte chunks per lane at kbase and kbase+16.
// Works for both global and LDS-backed pointers (inlined; addrspace inferred).
__device__ __forceinline__ v16bf ld_frag(const __bf16* base, int ld, int rc,
                                         int k0, int lane) {
  const int kb = k0 + ((lane >> 4) << 3);
  const __bf16* p = base + (size_t)rc * (size_t)ld + kb;
  union { v16bf v; i4v q[2]; } u;
  u.q[0] = *(const i4v*)(p);        // k = kb .. kb+7
  u.q[1] = *(const i4v*)(p + 16);   // k = kb+16 .. kb+23
  return u.v;
}

__device__ __forceinline__ v8f wmma_bf16(v16bf a, v16bf b, v8f c) {
  // (neg_a, A, neg_b, B, c_mod, C, reuse_a, reuse_b)
  return __builtin_amdgcn_wmma_f32_16x16x32_bf16(false, a, false, b,
                                                 (short)0, c, false, false);
}

__device__ __forceinline__ float sigmoidf_(float x) {
  return 1.0f / (1.0f + __expf(-x));
}

// ---------------------------------------------------------------------------
// Elementwise utility kernels
// ---------------------------------------------------------------------------
__global__ __launch_bounds__(256) void k_cvt_bf16(const float* __restrict__ s,
                                                  __bf16* __restrict__ d, int n) {
  for (int i = blockIdx.x * 256 + threadIdx.x; i < n; i += gridDim.x * 256)
    d[i] = (__bf16)s[i];
}

__global__ __launch_bounds__(256) void k_zero_u32(unsigned* __restrict__ p, int n) {
  for (int i = blockIdx.x * 256 + threadIdx.x; i < n; i += gridDim.x * 256)
    p[i] = 0u;
}

// ---------------------------------------------------------------------------
// Big GEMM: Cout[M,N] = A[M,K] * W[N,K]^T + bias0[N] + bias1[N]   (fp32 out)
// 2x2 register tiling: each wave computes a 32x32 block (4 WMMA tiles) so
// each K-step needs 2 A-frags + 2 B-frags for 4 WMMAs (32 B/lane per WMMA).
// M, N multiples of 32; K multiple of 32.
// ---------------------------------------------------------------------------
__global__ __launch_bounds__(256) void k_gemm_bias(
    const __bf16* __restrict__ A, const __bf16* __restrict__ W,
    const float* __restrict__ bias0, const float* __restrict__ bias1,
    float* __restrict__ Cout, int M, int N, int K) {
  const int lane = threadIdx.x & 31;
  const int nsupN = N >> 5;
  const int nsup  = (M >> 5) * nsupN;
  const int nwaves = (gridDim.x * blockDim.x) >> 5;

  for (int sup = (blockIdx.x * blockDim.x + threadIdx.x) >> 5; sup < nsup;
       sup += nwaves) {
    const int m0 = (sup / nsupN) << 5;
    const int n0 = (sup % nsupN) << 5;
    const int nn0 = n0 + (lane & 15);
    const int nn1 = nn0 + 16;
    const int ma0 = m0 + (lane & 15);
    const int ma1 = ma0 + 16;
    const float bi0 = bias0[nn0] + bias1[nn0];
    const float bi1 = bias0[nn1] + bias1[nn1];

    v8f acc00, acc01, acc10, acc11;
#pragma unroll
    for (int r = 0; r < 8; ++r) {
      acc00[r] = bi0; acc01[r] = bi1; acc10[r] = bi0; acc11[r] = bi1;
    }

    for (int k0 = 0; k0 < K; k0 += 32) {
      v16bf a0 = ld_frag(A, K, ma0, k0, lane);
      v16bf a1 = ld_frag(A, K, ma1, k0, lane);
      v16bf b0 = ld_frag(W, K, nn0, k0, lane);
      v16bf b1 = ld_frag(W, K, nn1, k0, lane);
      acc00 = wmma_bf16(a0, b0, acc00);
      acc01 = wmma_bf16(a0, b1, acc01);
      acc10 = wmma_bf16(a1, b0, acc10);
      acc11 = wmma_bf16(a1, b1, acc11);
    }

    const int mlo = (lane >> 4) << 3;      // f32 C/D layout: m = r + 8*(lane>>4)
#pragma unroll
    for (int r = 0; r < 8; ++r) {
      const size_t r0 = (size_t)(m0 + mlo + r) * (size_t)N;
      const size_t r1 = (size_t)(m0 + 16 + mlo + r) * (size_t)N;
      Cout[r0 + nn0] = acc00[r];
      Cout[r0 + nn1] = acc01[r];
      Cout[r1 + nn0] = acc10[r];
      Cout[r1 + nn1] = acc11[r];
    }
  }
}

// ---------------------------------------------------------------------------
// One LSTM timestep.
//   gates = xW[b, t, :] + h_prev @ W_hh^T ; then elementwise cell update.
// Block layout: 32 blocks x 256 threads (8 waves). Each block owns one
// 16-row batch tile of h_prev (32 KB), staged into LDS once via the CDNA5
// async-to-LDS path, and 8 hidden-column tiles (one per wave). Each wave
// computes all 4 gate tiles for its column tile (A-fragment reused 4x) and
// finishes the LSTM nonlinearity in-register on the f32 accumulators.
// ---------------------------------------------------------------------------
__global__ __launch_bounds__(256) void k_lstm_step(
    const float* __restrict__ xW,          // [B*T, 4H] fp32, rows b*T+t
    int t,
    const __bf16* __restrict__ Whh,        // [4H, H] bf16
    const __bf16* __restrict__ hprev,      // [B, H] bf16
    __bf16* __restrict__ hnext,            // [B, H] bf16
    float* __restrict__ cst,               // [B, H] fp32 (in/out)
    __bf16* __restrict__ hseq) {           // [B*T, H] bf16 or nullptr
  __shared__ __bf16 shA[16 * HH];          // 32 KB: this block's h_prev rows

  const int lane = threadIdx.x & 31;
  const int wave = threadIdx.x >> 5;       // 0..7
  const int mt = blockIdx.x & 3;           // batch tile (B=64 -> 4 tiles)
  const int ht = ((blockIdx.x >> 2) << 3) + wave;  // hidden-col tile 0..63
  const int m0 = mt << 4;
  const int n0 = ht << 4;
  const int nn  = n0 + (lane & 15);
  const int mlo = (lane >> 4) << 3;

  // --- stage h_prev[m0 .. m0+15][:] (contiguous 32 KB) into LDS ------------
  {
    const char* gsrc = (const char*)(hprev + (size_t)m0 * HH);
    char* ldst = (char*)shA;
#pragma unroll
    for (int i = 0; i < 8; ++i) {
      const int boff = (threadIdx.x + (i << 8)) << 4;   // 16 B per thread-iter
#if HAVE_ASYNC_LDS
      __builtin_amdgcn_global_load_async_to_lds_b128(
          (AS1 i4v*)(gsrc + boff), (AS3 i4v*)(ldst + boff), 0, 0);
#else
      *(i4v*)(ldst + boff) = *(const i4v*)(gsrc + boff);
#endif
    }
#if HAVE_ASYNC_LDS
#  if defined(__has_builtin) && __has_builtin(__builtin_amdgcn_s_wait_asynccnt)
    __builtin_amdgcn_s_wait_asynccnt(0);
#  else
    asm volatile("s_wait_asynccnt 0" ::: "memory");
#  endif
#endif
    __syncthreads();
  }

  // --- accumulators init from precomputed input projections ----------------
  v8f acc[4];
#pragma unroll
  for (int g = 0; g < 4; ++g) {
#pragma unroll
    for (int r = 0; r < 8; ++r) {
      const size_t row = (size_t)(m0 + mlo + r) * TT + t;
      acc[g][r] = xW[row * (size_t)G4H + (g * HH + nn)];
    }
  }

  // --- recurrent GEMM: A from LDS, W_hh streamed from L2 --------------------
  for (int k0 = 0; k0 < HH; k0 += 32) {
    v16bf af = ld_frag(shA, HH, lane & 15, k0, lane);
#pragma unroll
    for (int g = 0; g < 4; ++g) {
      v16bf bf = ld_frag(Whh, HH, g * HH + nn, k0, lane);
      acc[g] = wmma_bf16(af, bf, acc[g]);
    }
  }

  // --- LSTM cell update (i, f, g, o) ---------------------------------------
#pragma unroll
  for (int r = 0; r < 8; ++r) {
    const int m = m0 + mlo + r;
    const size_t idx = (size_t)m * HH + nn;
    const float iv = sigmoidf_(acc[0][r]);
    const float fv = sigmoidf_(acc[1][r]);
    const float gv = tanhf(acc[2][r]);
    const float ov = sigmoidf_(acc[3][r]);
    const float cn = fv * cst[idx] + iv * gv;
    cst[idx] = cn;
    const __bf16 hv = (__bf16)(ov * tanhf(cn));
    hnext[idx] = hv;
    if (hseq) hseq[((size_t)m * TT + t) * (size_t)HH + nn] = hv;
  }
}

// ---------------------------------------------------------------------------
// Dense head: out[B, C] = h_last[B, H] @ W_d[C, H]^T + b_d   (C = 1000)
// ---------------------------------------------------------------------------
__global__ __launch_bounds__(256) void k_dense(
    const __bf16* __restrict__ hlast, const __bf16* __restrict__ Wd,
    const float* __restrict__ bd, float* __restrict__ out) {
  const int lane  = threadIdx.x & 31;
  const int gwave = (blockIdx.x * blockDim.x + threadIdx.x) >> 5;
  const int ntN = (CC + 15) >> 4;          // 63 column tiles
  if (gwave >= 4 * ntN) return;            // wave-uniform guard (EXEC stays full)
  const int m0 = (gwave / ntN) << 4;
  const int n0 = (gwave % ntN) << 4;
  const int nn = n0 + (lane & 15);
  const bool valid = (nn < CC);
  const int nnc = valid ? nn : (CC - 1);   // clamp for safe loads; store masked
  const int ma  = m0 + (lane & 15);
  const int mlo = (lane >> 4) << 3;

  v8f acc;
  const float binit = bd[nnc];
#pragma unroll
  for (int r = 0; r < 8; ++r) acc[r] = binit;

  for (int k0 = 0; k0 < HH; k0 += 32) {
    v16bf af = ld_frag(hlast, HH, ma, k0, lane);
    v16bf bf = ld_frag(Wd, HH, nnc, k0, lane);
    acc = wmma_bf16(af, bf, acc);
  }
#pragma unroll
  for (int r = 0; r < 8; ++r)
    if (valid) out[(size_t)(m0 + mlo + r) * CC + nn] = acc[r];
}

// ---------------------------------------------------------------------------
// Host launcher
// ---------------------------------------------------------------------------
extern "C" void kernel_launch(void* const* d_in, const int* in_sizes, int n_in,
                              void* d_out, int out_size, void* d_ws, size_t ws_size,
                              hipStream_t stream) {
  (void)in_sizes; (void)n_in; (void)out_size; (void)ws_size;
  const float* x     = (const float*)d_in[0];
  const float* W_ih0 = (const float*)d_in[1];
  const float* W_hh0 = (const float*)d_in[2];
  const float* b_ih0 = (const float*)d_in[3];
  const float* b_hh0 = (const float*)d_in[4];
  const float* W_ih1 = (const float*)d_in[5];
  const float* W_hh1 = (const float*)d_in[6];
  const float* b_ih1 = (const float*)d_in[7];
  const float* b_hh1 = (const float*)d_in[8];
  const float* W_d   = (const float*)d_in[9];
  const float* b_d   = (const float*)d_in[10];
  float* out = (float*)d_out;

  // Workspace carve-up (256B aligned)
  char* ws = (char*)d_ws;
  size_t off = 0;
  auto alloc = [&](size_t bytes) -> void* {
    off = (off + 255) & ~(size_t)255;
    void* p = ws + off;
    off += bytes;
    return p;
  };
  __bf16* xb     = (__bf16*)alloc((size_t)BB * TT * II * 2);   // 32 MB
  __bf16* wih0b  = (__bf16*)alloc((size_t)G4H * II * 2);       // 4 MB
  __bf16* whh0b  = (__bf16*)alloc((size_t)G4H * HH * 2);       // 8 MB
  __bf16* wih1b  = (__bf16*)alloc((size_t)G4H * HH * 2);       // 8 MB
  __bf16* whh1b  = (__bf16*)alloc((size_t)G4H * HH * 2);       // 8 MB
  __bf16* wdb    = (__bf16*)alloc((size_t)CC * HH * 2);        // 2 MB
  float*  xw     = (float*) alloc((size_t)BB * TT * G4H * 4);  // 512 MB (reused L0/L1)
  __bf16* hseq   = (__bf16*)alloc((size_t)BB * TT * HH * 2);   // 64 MB
  __bf16* hA     = (__bf16*)alloc((size_t)BB * HH * 2);
  __bf16* hB     = (__bf16*)alloc((size_t)BB * HH * 2);
  float*  cbuf   = (float*) alloc((size_t)BB * HH * 4);

  // 1) fp32 -> bf16 conversions
  k_cvt_bf16<<<2048, 256, 0, stream>>>(x,     xb,    BB * TT * II);
  k_cvt_bf16<<<2048, 256, 0, stream>>>(W_ih0, wih0b, G4H * II);
  k_cvt_bf16<<<2048, 256, 0, stream>>>(W_hh0, whh0b, G4H * HH);
  k_cvt_bf16<<<2048, 256, 0, stream>>>(W_ih1, wih1b, G4H * HH);
  k_cvt_bf16<<<2048, 256, 0, stream>>>(W_hh1, whh1b, G4H * HH);
  k_cvt_bf16<<<2048, 256, 0, stream>>>(W_d,   wdb,   CC * HH);

  // 2) layer 0 input projections: xw = x @ W_ih0^T + b_ih0 + b_hh0
  k_gemm_bias<<<4096, 256, 0, stream>>>(xb, wih0b, b_ih0, b_hh0, xw,
                                        BB * TT, G4H, II);

  // 3) layer 0 recurrent scan
  k_zero_u32<<<64, 256, 0, stream>>>((unsigned*)hA,   BB * HH / 2);
  k_zero_u32<<<64, 256, 0, stream>>>((unsigned*)cbuf, BB * HH);
  __bf16 *prev = hA, *next = hB;
  for (int t = 0; t < TT; ++t) {
    k_lstm_step<<<32, 256, 0, stream>>>(xw, t, whh0b, prev, next, cbuf, hseq);
    __bf16* tmp = prev; prev = next; next = tmp;
  }

  // 4) layer 1 input projections (reuse xw): xw = hseq @ W_ih1^T + b_ih1 + b_hh1
  k_gemm_bias<<<4096, 256, 0, stream>>>(hseq, wih1b, b_ih1, b_hh1, xw,
                                        BB * TT, G4H, HH);

  // 5) layer 1 recurrent scan (only final h needed)
  k_zero_u32<<<64, 256, 0, stream>>>((unsigned*)hA,   BB * HH / 2);
  k_zero_u32<<<64, 256, 0, stream>>>((unsigned*)cbuf, BB * HH);
  prev = hA; next = hB;
  for (int t = 0; t < TT; ++t) {
    k_lstm_step<<<32, 256, 0, stream>>>(xw, t, whh1b, prev, next, cbuf, nullptr);
    __bf16* tmp = prev; prev = next; next = tmp;
  }

  // 6) dense head on h_last (in `prev` after the loop)
  k_dense<<<32, 256, 0, stream>>>(prev, wdb, b_d, out);
}
